// GraphAttentionLayer_18451179503687
// MI455X (gfx1250) — matrix operand order repaired and checked by reading, not compiled, not run on hardware
//
#include <hip/hip_runtime.h>
#include <hip/hip_bf16.h>

typedef __attribute__((ext_vector_type(16))) _Float16 v16h;
typedef __attribute__((ext_vector_type(8)))  float    v8f;

#define NB    8
#define NN    2048
#define FIN   512
#define FOUT  256
#define HID   64
#define ALPHA 0.2f

#define WMMA_F16(A, B, C) \
  __builtin_amdgcn_wmma_f32_16x16x32_f16(false, (A), false, (B), (short)0, (C), false, false)

// ---------------------------------------------------------------------------
// Prep: wT[c][k] = weight[k][c] as f16 ; v1 = ht@a[:64], v2 = ht@a[64:]
// ---------------------------------------------------------------------------
__global__ void gat_prep(const float* __restrict__ weight,  // [FIN][FOUT]
                         const float* __restrict__ ht,      // [FOUT][HID]
                         const float* __restrict__ a,       // [2*HID]
                         _Float16* __restrict__ wT,         // [FOUT][FIN]
                         float* __restrict__ v1,
                         float* __restrict__ v2) {
  int t = blockIdx.x * blockDim.x + threadIdx.x;
  if (t < FIN * FOUT) {
    int k = t / FOUT, c = t % FOUT;
    wT[(size_t)c * FIN + k] = (_Float16)weight[t];
  }
  if (t < FOUT) {
    float s1 = 0.f, s2 = 0.f;
    for (int k = 0; k < HID; ++k) {
      float hv = ht[t * HID + k];
      s1 = fmaf(hv, a[k], s1);
      s2 = fmaf(hv, a[HID + k], s2);
    }
    v1[t] = s1;
    v2[t] = s2;
  }
}

// ---------------------------------------------------------------------------
// GEMM1: h16 = input @ weight  (M=16384, K=512, N=256), f16 WMMA, f32 acc.
// Double-buffered LDS tiles, one-step-ahead A staging,
// two-live-buffer interleaved B-fragment fetch.
// ---------------------------------------------------------------------------
__global__ __launch_bounds__(256) void gat_gemm1(
    const float*    __restrict__ inp,   // [16384][512] f32
    const _Float16* __restrict__ wT,    // [256][512]   f16
    _Float16*       __restrict__ h16) { // [16384][256] f16
  __shared__ __align__(64) _Float16 Bt[2][FOUT * 32];  // col-major: [c*32 + k]

  const int lane = threadIdx.x & 31;
  const int w    = threadIdx.x >> 5;
  const int m0   = blockIdx.x * 128 + w * 16;
  const int mrow = m0 + (lane & 15);
  const int off  = (lane < 16) ? 0 : 16;
  const int koff = (lane < 16) ? 0 : 16;
  const int col  = lane & 15;
  const int NK   = FIN / 32;

  const uint4* wrow = (const uint4*)(wT + (size_t)threadIdx.x * FIN);
  const float* arow = inp + (size_t)mrow * FIN;

  // preload: B tile 0 -> LDS buf0, A raw for k-step 0
  {
    uint4* dst = (uint4*)(Bt[0] + threadIdx.x * 32);
    dst[0] = wrow[0]; dst[1] = wrow[1]; dst[2] = wrow[2]; dst[3] = wrow[3];
  }
  float4 a0 = *(const float4*)(arow + off);
  float4 a1 = *(const float4*)(arow + off + 4);
  float4 a2 = *(const float4*)(arow + off + 8);
  float4 a3 = *(const float4*)(arow + off + 12);

  v8f acc[16] = {};
  __syncthreads();

  for (int kk = 0; kk < NK; ++kk) {
    const int cur   = kk & 1;
    const int nxt   = cur ^ 1;
    const bool more = (kk + 1) < NK;

    uint4  s0, s1, s2, s3;   // staged B tile (global -> reg)
    float4 n0, n1, n2, n3;   // staged A raw for next step
    if (more) {
      const uint4* src = wrow + (kk + 1) * 4;
      s0 = src[0]; s1 = src[1]; s2 = src[2]; s3 = src[3];
      const float* ap = arow + (kk + 1) * 32;
      n0 = *(const float4*)(ap + off);
      n1 = *(const float4*)(ap + off + 4);
      n2 = *(const float4*)(ap + off + 8);
      n3 = *(const float4*)(ap + off + 12);
    }

    // convert current A raw -> f16 fragment
    v16h av;
    av[0]  = (_Float16)a0.x; av[1]  = (_Float16)a0.y;
    av[2]  = (_Float16)a0.z; av[3]  = (_Float16)a0.w;
    av[4]  = (_Float16)a1.x; av[5]  = (_Float16)a1.y;
    av[6]  = (_Float16)a1.z; av[7]  = (_Float16)a1.w;
    av[8]  = (_Float16)a2.x; av[9]  = (_Float16)a2.y;
    av[10] = (_Float16)a2.z; av[11] = (_Float16)a2.w;
    av[12] = (_Float16)a3.x; av[13] = (_Float16)a3.y;
    av[14] = (_Float16)a3.z; av[15] = (_Float16)a3.w;

    // two-live-buffer interleaved B-fragment fetch + 16 WMMAs
    {
      const _Float16* base = Bt[cur] + col * 32 + koff;
      v16h b0 = *(const v16h*)(base);
      v16h b1 = *(const v16h*)(base + 16 * 32);
      #pragma unroll
      for (int t = 0; t < 14; t += 2) {
        acc[t] = WMMA_F16(av, b0, acc[t]);
        b0 = *(const v16h*)(base + (t + 2) * 16 * 32);
        acc[t + 1] = WMMA_F16(av, b1, acc[t + 1]);
        b1 = *(const v16h*)(base + (t + 3) * 16 * 32);
      }
      acc[14] = WMMA_F16(av, b0, acc[14]);
      acc[15] = WMMA_F16(av, b1, acc[15]);
    }

    if (more) {
      uint4* dst = (uint4*)(Bt[nxt] + threadIdx.x * 32);
      dst[0] = s0; dst[1] = s1; dst[2] = s2; dst[3] = s3;
      a0 = n0; a1 = n1; a2 = n2; a3 = n3;
    }
    __syncthreads();
  }

  const int rbase = (lane < 16) ? 0 : 8;
  #pragma unroll
  for (int t = 0; t < 16; ++t) {
    #pragma unroll
    for (int r = 0; r < 8; ++r) {
      h16[(size_t)(m0 + r + rbase) * FOUT + 16 * t + col] = (_Float16)acc[t][r];
    }
  }
}

// ---------------------------------------------------------------------------
// Transpose h16 [16384][256] -> hT [256][16384] (f16) via LDS tiles
// ---------------------------------------------------------------------------
__global__ void gat_transpose(const _Float16* __restrict__ h16,
                              _Float16* __restrict__ hT) {
  __shared__ _Float16 tile[32][33];
  const int tx = threadIdx.x & 31;
  const int ty = threadIdx.x >> 5;       // 0..7
  const int m0 = blockIdx.x * 32;        // over 16384
  const int c0 = blockIdx.y * 32;        // over 256
  #pragma unroll
  for (int i = 0; i < 4; ++i) {
    int r = ty + 8 * i;
    tile[r][tx] = h16[(size_t)(m0 + r) * FOUT + c0 + tx];
  }
  __syncthreads();
  #pragma unroll
  for (int i = 0; i < 4; ++i) {
    int c = ty + 8 * i;
    hT[(size_t)(c0 + c) * (NB * NN) + m0 + tx] = tile[tx][c];
  }
}

// ---------------------------------------------------------------------------
// f1[i] = h[i]@v1, f2[i] = h[i]@v2   (reads hT column-wise: coalesced)
// ---------------------------------------------------------------------------
__global__ void gat_f12(const _Float16* __restrict__ hT,
                        const float* __restrict__ v1,
                        const float* __restrict__ v2,
                        float* __restrict__ f1,
                        float* __restrict__ f2) {
  int i = blockIdx.x * blockDim.x + threadIdx.x;  // 0..16383
  float s1 = 0.f, s2 = 0.f;
  for (int c = 0; c < FOUT; ++c) {
    float hv = (float)hT[(size_t)c * (NB * NN) + i];
    s1 = fmaf(hv, v1[c], s1);
    s2 = fmaf(hv, v2[c], s2);
  }
  f1[i] = s1;
  f2[i] = s2;
}

// ---------------------------------------------------------------------------
// Per-batch max over f2 (for the safe softmax shift)
// ---------------------------------------------------------------------------
__global__ void gat_m2(const float* __restrict__ f2, float* __restrict__ m2) {
  __shared__ float red[256];
  const int b = blockIdx.x;
  float m = -1e30f;
  for (int j = threadIdx.x; j < NN; j += 256) m = fmaxf(m, f2[b * NN + j]);
  red[threadIdx.x] = m;
  __syncthreads();
  for (int s = 128; s > 0; s >>= 1) {
    if (threadIdx.x < s) red[threadIdx.x] = fmaxf(red[threadIdx.x], red[threadIdx.x + s]);
    __syncthreads();
  }
  if (threadIdx.x == 0) m2[b] = red[0];
}

// ---------------------------------------------------------------------------
// Fused masked-softmax attention + h_prime + ELU, fully pipelined:
//  - LDS h-tile double buffered (stage next tile during WMMA block)
//  - adj/f2 raw loads issued BEFORE the WMMA block, exp chain after it
//  - two-live-buffer interleaved LDS B-fragment fetch
// p_mj = exp(leaky(f1[m]+f2[j]) - shift_m) if adj>0 else 0  (shift_m => p<=1)
// out = elu( (P @ H) / rowsum )
// ---------------------------------------------------------------------------
struct PRaw {
  float4 fa0, fa1, fb0, fb1;
  int4   d0, d1, d2, d3;
};

__global__ __launch_bounds__(256) void gat_attn(
    const int*      __restrict__ adj,   // [NB][NN][NN]
    const _Float16* __restrict__ hT,    // [FOUT][NB*NN]
    const float*    __restrict__ f1,
    const float*    __restrict__ f2,
    const float*    __restrict__ m2,    // [NB]
    float*          __restrict__ out) { // [NB][NN][FOUT]
  __shared__ __align__(64) _Float16 Bt[2][FOUT * 32];
  __shared__ float rowsum[8 * 16];

  const int lane  = threadIdx.x & 31;
  const int w     = threadIdx.x >> 5;
  const int b     = blockIdx.x >> 4;
  const int m0    = (blockIdx.x & 15) * 128 + w * 16;
  const int myrow = m0 + (lane & 15);
  const int off   = (lane < 16) ? 0 : 16;
  const int koff  = (lane < 16) ? 0 : 16;
  const int col   = lane & 15;
  const int NJ    = NN / 32;

  const float  f1m    = f1[b * NN + myrow];
  const float  shift  = fmaxf(0.f, f1m + m2[b]);
  const float* f2b    = f2 + b * NN;
  const int*   adjrow = adj + ((size_t)b * NN + myrow) * NN;
  const uint4* hrow   = (const uint4*)(hT + (size_t)threadIdx.x * (NB * NN) + b * NN);

  v8f   acc[16] = {};
  float ssum    = 0.f;

  auto loadP = [&](int j0) -> PRaw {
    PRaw r;
    r.fa0 = *(const float4*)(f2b + j0 + off);
    r.fa1 = *(const float4*)(f2b + j0 + off + 4);
    r.fb0 = *(const float4*)(f2b + j0 + off + 8);
    r.fb1 = *(const float4*)(f2b + j0 + off + 12);
    r.d0  = *(const int4*)(adjrow + j0 + off);
    r.d1  = *(const int4*)(adjrow + j0 + off + 4);
    r.d2  = *(const int4*)(adjrow + j0 + off + 8);
    r.d3  = *(const int4*)(adjrow + j0 + off + 12);
    return r;
  };
  auto makeP = [&](const PRaw& r) -> v16h {
    float fv[16] = {r.fa0.x, r.fa0.y, r.fa0.z, r.fa0.w,
                    r.fa1.x, r.fa1.y, r.fa1.z, r.fa1.w,
                    r.fb0.x, r.fb0.y, r.fb0.z, r.fb0.w,
                    r.fb1.x, r.fb1.y, r.fb1.z, r.fb1.w};
    int   mv[16] = {r.d0.x, r.d0.y, r.d0.z, r.d0.w,
                    r.d1.x, r.d1.y, r.d1.z, r.d1.w,
                    r.d2.x, r.d2.y, r.d2.z, r.d2.w,
                    r.d3.x, r.d3.y, r.d3.z, r.d3.w};
    v16h p;
    #pragma unroll
    for (int i = 0; i < 16; ++i) {
      float e = f1m + fv[i];
      e = (e >= 0.f) ? e : ALPHA * e;
      float pe = (mv[i] > 0) ? __expf(e - shift) : 0.f;
      ssum += pe;
      p[i] = (_Float16)pe;
    }
    return p;
  };

  // preload: tile 0 -> LDS buf0; P fragment for step 0
  {
    uint4* dst = (uint4*)(Bt[0] + threadIdx.x * 32);
    dst[0] = hrow[0]; dst[1] = hrow[1]; dst[2] = hrow[2]; dst[3] = hrow[3];
  }
  v16h pv = makeP(loadP(0));
  __syncthreads();

  for (int jj = 0; jj < NJ; ++jj) {
    const int cur   = jj & 1;
    const int nxt   = cur ^ 1;
    const bool more = (jj + 1) < NJ;

    PRaw  praw;
    uint4 s0, s1, s2, s3;
    if (more) {
      praw = loadP((jj + 1) * 32);        // adj/f2 loads in flight during WMMAs
      const uint4* src = hrow + (jj + 1) * 4;
      s0 = src[0]; s1 = src[1]; s2 = src[2]; s3 = src[3];
    }

    // two-live-buffer interleaved B-fragment fetch + 16 WMMAs
    {
      const _Float16* base = Bt[cur] + col * 32 + koff;
      v16h b0 = *(const v16h*)(base);
      v16h b1 = *(const v16h*)(base + 16 * 32);
      #pragma unroll
      for (int t = 0; t < 14; t += 2) {
        acc[t] = WMMA_F16(pv, b0, acc[t]);
        b0 = *(const v16h*)(base + (t + 2) * 16 * 32);
        acc[t + 1] = WMMA_F16(pv, b1, acc[t + 1]);
        b1 = *(const v16h*)(base + (t + 3) * 16 * 32);
      }
      acc[14] = WMMA_F16(pv, b0, acc[14]);
      acc[15] = WMMA_F16(pv, b1, acc[15]);
    }

    if (more) {
      pv = makeP(praw);                   // exp chain after the WMMA block
      uint4* dst = (uint4*)(Bt[nxt] + threadIdx.x * 32);
      dst[0] = s0; dst[1] = s1; dst[2] = s2; dst[3] = s3;
    }
    __syncthreads();
  }

  // combine the two K-halves of the row sum, publish per-row denominators
  ssum += __shfl_xor(ssum, 16, 32);
  if (lane < 16) rowsum[w * 16 + lane] = ssum;
  __syncthreads();

  const int rbase = (lane < 16) ? 0 : 8;
  #pragma unroll
  for (int t = 0; t < 16; ++t) {
    #pragma unroll
    for (int r = 0; r < 8; ++r) {
      float s = rowsum[w * 16 + r + rbase];
      float v = acc[t][r] / s;
      v = (v > 0.f) ? v : (__expf(v) - 1.f);  // ELU (alpha=1)
      out[((size_t)b * NN + m0 + r + rbase) * FOUT + 16 * t + col] = v;
    }
  }
}

// ---------------------------------------------------------------------------
extern "C" void kernel_launch(void* const* d_in, const int* in_sizes, int n_in,
                              void* d_out, int out_size, void* d_ws, size_t ws_size,
                              hipStream_t stream) {
  const float* inp    = (const float*)d_in[0];  // [8][2048][512]
  const int*   adj    = (const int*)  d_in[1];  // [8][2048][2048]
  const float* weight = (const float*)d_in[2];  // [512][256]
  const float* ht     = (const float*)d_in[3];  // [256][64]
  const float* a      = (const float*)d_in[4];  // [128]
  float* out = (float*)d_out;

  char*  ws = (char*)d_ws;
  size_t o  = 0;
  auto alloc = [&](size_t bytes) -> void* {
    void* p = ws + o;
    o = (o + bytes + 255) & ~(size_t)255;
    return p;
  };
  _Float16* wT  = (_Float16*)alloc((size_t)FIN * FOUT * 2);
  float*    v1  = (float*)   alloc(FOUT * 4);
  float*    v2  = (float*)   alloc(FOUT * 4);
  float*    m2  = (float*)   alloc(NB * 4);
  float*    f1  = (float*)   alloc((size_t)NB * NN * 4);
  float*    f2  = (float*)   alloc((size_t)NB * NN * 4);
  _Float16* h16 = (_Float16*)alloc((size_t)NB * NN * FOUT * 2);
  _Float16* hT  = (_Float16*)alloc((size_t)NB * NN * FOUT * 2);

  gat_prep<<<dim3((FIN * FOUT + 255) / 256), dim3(256), 0, stream>>>(
      weight, ht, a, wT, v1, v2);
  gat_gemm1<<<dim3(128), dim3(256), 0, stream>>>(inp, wT, h16);
  gat_transpose<<<dim3((NB * NN) / 32, FOUT / 32), dim3(256), 0, stream>>>(h16, hT);
  gat_f12<<<dim3((NB * NN) / 256), dim3(256), 0, stream>>>(hT, v1, v2, f1, f2);
  gat_m2<<<dim3(NB), dim3(256), 0, stream>>>(f2, m2);
  gat_attn<<<dim3(128), dim3(256), 0, stream>>>(adj, hT, f1, f2, m2, out);
}
